// ScaledDotProductAttention_50577534878008
// MI455X (gfx1250) — matrix operand order repaired and checked by reading, not compiled, not run on hardware
//
#include <hip/hip_runtime.h>
#include <hip/hip_bf16.h>

typedef __attribute__((ext_vector_type(16))) _Float16 v16h;
typedef __attribute__((ext_vector_type(8)))  _Float16 v8h;
typedef __attribute__((ext_vector_type(8)))  float    v8f;

constexpr int Bc = 2, Hc = 16, Sc = 2048, DKc = 64;
constexpr float NEGV = -1e9f;
constexpr int QTILE = 128;    // q rows per block (16 per wave, 8 waves)
constexpr int KTILE = 64;     // keys per K-step
constexpr int NWAVE = 8;      // 256 threads, wave32
constexpr int NSTEP = Sc / KTILE;
constexpr int STG   = KTILE * DKc / 256;  // 16 staged elements per thread

__device__ __forceinline__ v16h combine8(v8h lo, v8h hi) {
  v16h r;
#pragma unroll
  for (int i = 0; i < 8; ++i) { r[i] = lo[i]; r[i + 8] = hi[i]; }
  return r;
}

__global__ __launch_bounds__(256)
void fa_fused_kernel(const float* __restrict__ Q, const float* __restrict__ K,
                     const float* __restrict__ V,
                     const unsigned char* __restrict__ mask,   // jnp.bool_ -> 1 byte
                     const float* __restrict__ mat,
                     float* __restrict__ out)
{
  __shared__ __align__(32) _Float16 Qs[QTILE * DKc];          // 16 KB
  __shared__ __align__(32) _Float16 Ks[KTILE * DKc];          //  8 KB [key][dim]
  __shared__ __align__(32) _Float16 VsT[DKc * KTILE];         //  8 KB [dim][key]
  __shared__ __align__(32) _Float16 Pws[NWAVE][16 * KTILE];   // 16 KB per-wave P scratch

  const int tid  = threadIdx.x;
  const int wave = tid >> 5;
  const int lane = tid & 31;
  const int hi   = lane >> 4;   // half-wave select (K/M split per ISA layouts)
  const int ln   = lane & 15;

  const int qtiles = Sc / QTILE;
  const int bh = blockIdx.x / qtiles;
  const int q0 = (blockIdx.x % qtiles) * QTILE;

  const size_t headD = (size_t)bh * Sc * DKc;   // base of this head in Q/K/V/out
  const size_t headS = (size_t)bh * Sc * Sc;    // base of this head in matrix/mask

  // ---- pipeline prologue: issue global loads of K/V tile 0 into registers ----
  float kst[STG], vst[STG];
  {
    const float* Kt = K + headD;
    const float* Vt = V + headD;
#pragma unroll
    for (int i = 0; i < STG; ++i) {
      int idx = tid + i * 256;
      kst[i] = Kt[idx];
      vst[i] = Vt[idx];
    }
  }

  // ---- stage Q tile (fp32 -> f16) into LDS ----
  const float* Qp = Q + headD + (size_t)q0 * DKc;
#pragma unroll
  for (int i = 0; i < QTILE * DKc / 256; ++i) {
    int idx = tid + i * 256;
    Qs[idx] = (_Float16)Qp[idx];
  }
  __syncthreads();

  // ---- per-wave Q A-fragments (16 rows x 64 dims), held in VGPRs all kernel ----
  // A 16x32 f16 layout: lanes0-15: K 0..7 then 16..23; lanes16-31: K 8..15 then 24..31.
  const int arow = wave * 16 + ln;
  v16h qa[2];
#pragma unroll
  for (int s = 0; s < 2; ++s) {
    const _Float16* p0 = Qs + arow * DKc + (32 * s + hi * 8);
    qa[s] = combine8(*(const v8h*)p0, *(const v8h*)(p0 + 16));
  }

  const v8f vzero = {};
  v8f acc[4];
#pragma unroll
  for (int t = 0; t < 4; ++t) acc[t] = vzero;
  float mrow[8], lrow[8];
#pragma unroll
  for (int v = 0; v < 8; ++v) { mrow[v] = -__builtin_inff(); lrow[v] = 0.f; }

  const float scale = 0.125f;  // 1/sqrt(64)

  for (int kb = 0; kb < NSTEP; ++kb) {
    const int k0 = kb * KTILE;
    __syncthreads();  // all waves done reading previous Ks/VsT

    // ---- dump staged tile kb to LDS (f32->f16, V transposed) ----
#pragma unroll
    for (int i = 0; i < STG; ++i) {
      int idx = tid + i * 256;
      int ky = idx >> 6, dm = idx & 63;
      Ks[ky * DKc + dm]    = (_Float16)kst[i];
      VsT[dm * KTILE + ky] = (_Float16)vst[i];
    }

    // ---- issue global loads for tile kb+1; they drain during compute below ----
    if (kb + 1 < NSTEP) {
      const float* Kt = K + headD + (size_t)(k0 + KTILE) * DKc;
      const float* Vt = V + headD + (size_t)(k0 + KTILE) * DKc;
#pragma unroll
      for (int i = 0; i < STG; ++i) {
        int idx = tid + i * 256;
        kst[i] = Kt[idx];
        vst[i] = Vt[idx];
      }
    }
    __syncthreads();

    // ---- scores S = Q K^T : 16x64 per wave, 8 WMMAs ----
    v8f sc[4];
#pragma unroll
    for (int t = 0; t < 4; ++t) {
      v8f c = vzero;
#pragma unroll
      for (int s = 0; s < 2; ++s) {
        // B 32x16 layout: lane N = 16t+ln; dims 32s + hi*8 + 0..7 then +16..23.
        const _Float16* bp = Ks + (16 * t + ln) * DKc + 32 * s + hi * 8;
        v16h bk = combine8(*(const v8h*)bp, *(const v8h*)(bp + 16));
        c = __builtin_amdgcn_wmma_f32_16x16x32_f16(false, qa[s], false, bk,
                                                   (short)0, c, false, false);
      }
      sc[t] = c;
    }

    // ---- modulate: *(1/8)*matrix, mask -> NEG (non-temporal streams) ----
    float tmax[8];
#pragma unroll
    for (int v = 0; v < 8; ++v) tmax[v] = -__builtin_inff();
#pragma unroll
    for (int t = 0; t < 4; ++t) {
#pragma unroll
      for (int v = 0; v < 8; ++v) {
        size_t off = headS + (size_t)(q0 + wave * 16 + 8 * hi + v) * Sc
                   + (size_t)(k0 + 16 * t + ln);
        float m = __builtin_nontemporal_load(mat + off);
        unsigned char bm = __builtin_nontemporal_load(mask + off);
        float x = sc[t][v] * scale * m;
        x = bm ? NEGV : x;
        sc[t][v] = x;
        tmax[v] = fmaxf(tmax[v], x);
      }
    }

    // ---- row max across the 16-lane N group ----
#pragma unroll
    for (int v = 0; v < 8; ++v) {
#pragma unroll
      for (int off = 8; off >= 1; off >>= 1)
        tmax[v] = fmaxf(tmax[v], __shfl_xor(tmax[v], off, 32));
    }

    // ---- online softmax update ----
    float alpha[8], rsum[8];
#pragma unroll
    for (int v = 0; v < 8; ++v) {
      float mn = fmaxf(mrow[v], tmax[v]);
      alpha[v] = __expf(mrow[v] - mn);
      mrow[v] = mn;
      rsum[v] = 0.f;
    }
#pragma unroll
    for (int t = 0; t < 4; ++t) {
#pragma unroll
      for (int v = 0; v < 8; ++v) {
        float p = __expf(sc[t][v] - mrow[v]);
        rsum[v] += p;
        Pws[wave][(v + 8 * hi) * KTILE + 16 * t + ln] = (_Float16)p;
      }
    }
#pragma unroll
    for (int v = 0; v < 8; ++v) {
#pragma unroll
      for (int off = 8; off >= 1; off >>= 1)
        rsum[v] += __shfl_xor(rsum[v], off, 32);
      lrow[v] = alpha[v] * lrow[v] + rsum[v];
    }
#pragma unroll
    for (int t = 0; t < 4; ++t)
#pragma unroll
      for (int v = 0; v < 8; ++v)
        acc[t][v] *= alpha[v];

    // ---- O += P @ V : 8 WMMAs (P re-read in A layout from own wave's LDS scratch) ----
#pragma unroll
    for (int s = 0; s < 2; ++s) {
      const _Float16* ap0 = &Pws[wave][ln * KTILE + 32 * s + hi * 8];
      v16h ap = combine8(*(const v8h*)ap0, *(const v8h*)(ap0 + 16));
#pragma unroll
      for (int t = 0; t < 4; ++t) {
        // B 32x16: lane N = head dim 16t+ln; K keys 32s + hi*8 + 0..7 then +16..23 in VsT row
        const _Float16* bp = VsT + (16 * t + ln) * KTILE + 32 * s + hi * 8;
        v16h bv = combine8(*(const v8h*)bp, *(const v8h*)(bp + 16));
        acc[t] = __builtin_amdgcn_wmma_f32_16x16x32_f16(false, ap, false, bv,
                                                        (short)0, acc[t], false, false);
      }
    }
  }

  // ---- epilogue: normalize and store context ----
#pragma unroll
  for (int v = 0; v < 8; ++v) {
    float inv = 1.f / lrow[v];
#pragma unroll
    for (int t = 0; t < 4; ++t) {
      size_t off = headD + (size_t)(q0 + wave * 16 + 8 * hi + v) * DKc
                 + (size_t)(16 * t + ln);
      out[off] = acc[t][v] * inv;
    }
  }
}

extern "C" void kernel_launch(void* const* d_in, const int* in_sizes, int n_in,
                              void* d_out, int out_size, void* d_ws, size_t ws_size,
                              hipStream_t stream) {
  (void)in_sizes; (void)n_in; (void)out_size; (void)d_ws; (void)ws_size;
  const float* Q = (const float*)d_in[0];
  const float* K = (const float*)d_in[1];
  const float* V = (const float*)d_in[2];
  const unsigned char* mask = (const unsigned char*)d_in[3];
  const float* mat = (const float*)d_in[4];
  float* out = (float*)d_out;

  dim3 grid(Bc * Hc * (Sc / QTILE));  // 512 blocks
  dim3 block(256);                    // 8 wave32 waves
  fa_fused_kernel<<<grid, block, 0, stream>>>(Q, K, V, mask, mat, out);
}
